// MultiBoxLoss_89043261981361
// MI455X (gfx1250) — compile-verified, matrix operand order checked by reading, not run on hardware
//
#include <hip/hip_runtime.h>
#include <math.h>

#define B_   128
#define P_   8732
#define O_   16
#define C_   21
#define TPB  512
#define THRESH 0.5f
#define VAR0 0.1f
#define VAR1 0.2f

typedef __attribute__((ext_vector_type(16))) _Float16 v16h;
typedef __attribute__((ext_vector_type(8)))  float    v8f;

// One workgroup per image: matching + encoding + CE + hard-negative mining,
// all intermediates in LDS (320KB/WGP makes the 35KB CE vector trivial).
__global__ __launch_bounds__(TPB) void mbox_per_image(
    const float* __restrict__ loc_preds,   // [B,P,4]
    const float* __restrict__ conf_preds,  // [B,P,C]
    const float* __restrict__ boxes,       // [B,O,4] point form
    const int*   __restrict__ labels,      // [B,O]
    const float* __restrict__ priors,      // [P,4] center form
    float* __restrict__ ws_sl1, float* __restrict__ ws_cep,
    float* __restrict__ ws_neg, int* __restrict__ ws_npos)
{
    __shared__ float s_truth[O_][4];
    __shared__ float s_area[O_];
    __shared__ int   s_label[O_];
    __shared__ unsigned long long s_bestpt[O_];
    __shared__ int   s_force[O_];
    __shared__ float s_red[TPB];
    __shared__ float s_red2[TPB];
    __shared__ float s_wred[(TPB / 32) * 16];
    __shared__ float s_ce[P_];              // negative CE per prior (0 at positives)
    __shared__ int   s_cnt;
    __shared__ int   s_count;

    const int b   = blockIdx.x;
    const int tid = threadIdx.x;

    if (tid < O_ * 4) ((float*)s_truth)[tid] = boxes[(size_t)b * O_ * 4 + tid];
    if (tid < O_) { s_label[tid] = labels[b * O_ + tid]; s_bestpt[tid] = 0ULL; }
    __syncthreads();
    if (tid < O_)
        s_area[tid] = (s_truth[tid][2] - s_truth[tid][0]) * (s_truth[tid][3] - s_truth[tid][1]);
    __syncthreads();

    // ---- Pass 1: per-truth best prior via packed u64 max (tie -> smaller p) ----
    unsigned long long lb[O_];
#pragma unroll
    for (int o = 0; o < O_; ++o) lb[o] = 0ULL;

    for (int p = tid; p < P_; p += TPB) {
        float cx = priors[p*4+0], cy = priors[p*4+1], w = priors[p*4+2], h = priors[p*4+3];
        float bx0 = cx - w*0.5f, by0 = cy - h*0.5f, bx1 = cx + w*0.5f, by1 = cy + h*0.5f;
        float area_b = (bx1 - bx0) * (by1 - by0);
#pragma unroll
        for (int o = 0; o < O_; ++o) {
            float ltx = fmaxf(s_truth[o][0], bx0), lty = fmaxf(s_truth[o][1], by0);
            float rbx = fminf(s_truth[o][2], bx1), rby = fminf(s_truth[o][3], by1);
            float iw = fmaxf(rbx - ltx, 0.f), ih = fmaxf(rby - lty, 0.f);
            float inter = iw * ih;
            float iou = inter / (s_area[o] + area_b - inter);
            unsigned long long key =
                ((unsigned long long)__float_as_uint(iou) << 32) | (unsigned)(P_ - 1 - p);
            if (key > lb[o]) lb[o] = key;
        }
    }
#pragma unroll
    for (int o = 0; o < O_; ++o) atomicMax(&s_bestpt[o], lb[o]);
    __syncthreads();
    if (tid < O_) s_force[tid] = P_ - 1 - (int)(s_bestpt[tid] & 0xFFFFFFFFu);
    __syncthreads();

    // ---- Pass 2: matching result per prior + SmoothL1 + cross entropy ----
    float sl1_part = 0.f, cep_part = 0.f;
    int my_count = 0;
    for (int p = tid; p < P_; p += TPB) {
        float cx = priors[p*4+0], cy = priors[p*4+1], w = priors[p*4+2], h = priors[p*4+3];
        float bx0 = cx - w*0.5f, by0 = cy - h*0.5f, bx1 = cx + w*0.5f, by1 = cy + h*0.5f;
        float area_b = (bx1 - bx0) * (by1 - by0);
        float best_v = -1.f; int best_o = 0;
#pragma unroll
        for (int o = 0; o < O_; ++o) {
            float ltx = fmaxf(s_truth[o][0], bx0), lty = fmaxf(s_truth[o][1], by0);
            float rbx = fminf(s_truth[o][2], bx1), rby = fminf(s_truth[o][3], by1);
            float iw = fmaxf(rbx - ltx, 0.f), ih = fmaxf(rby - lty, 0.f);
            float inter = iw * ih;
            float iou = inter / (s_area[o] + area_b - inter);
            if (iou > best_v) { best_v = iou; best_o = o; }  // first-index tie like argmax
        }
#pragma unroll
        for (int o = 0; o < O_; ++o)                          // force-assign, last o wins
            if (s_force[o] == p) { best_o = o; best_v = 2.0f; }
        int conf = (best_v < THRESH) ? 0 : (s_label[best_o] + 1);

        // log-softmax CE over 21 classes (single read of the conf row)
        const float* cp = conf_preds + ((size_t)b * P_ + p) * C_;
        float m = cp[0];
#pragma unroll
        for (int c = 1; c < C_; ++c) m = fmaxf(m, cp[c]);
        float ssum = 0.f;
#pragma unroll
        for (int c = 0; c < C_; ++c) ssum += expf(cp[c] - m);
        float ce = (m + logf(ssum)) - cp[conf];

        if (conf > 0) {
            ++my_count;
            cep_part += ce;
            s_ce[p] = 0.f;
            // encode matched gt vs prior, SmoothL1 against prediction
            float tx0 = s_truth[best_o][0], ty0 = s_truth[best_o][1];
            float tx1 = s_truth[best_o][2], ty1 = s_truth[best_o][3];
            float g0 = ((tx0 + tx1) * 0.5f - cx) / (VAR0 * w);
            float g1 = ((ty0 + ty1) * 0.5f - cy) / (VAR0 * h);
            float g2 = logf((tx1 - tx0) / w) / VAR1;
            float g3 = logf((ty1 - ty0) / h) / VAR1;
            const float* lp = loc_preds + ((size_t)b * P_ + p) * 4;
            float d[4] = { lp[0]-g0, lp[1]-g1, lp[2]-g2, lp[3]-g3 };
#pragma unroll
            for (int i = 0; i < 4; ++i) {
                float ax = fabsf(d[i]);
                sl1_part += (ax < 1.f) ? 0.5f * ax * ax : ax - 0.5f;
            }
        } else {
            s_ce[p] = fmaxf(ce, 0.f);   // CE >= 0; clamp keeps bit-order monotone
        }
    }

    // ---- positive count via WMMA (exact: small integers, f32 accumulate) ----
    // D = ones(16x32) x splat(count)(32x16): every D element = 16*(c_n + c_{n+16})
    v16h aa, bb;
    _Float16 ch = (_Float16)(float)my_count;
#pragma unroll
    for (int i = 0; i < 16; ++i) { aa[i] = (_Float16)1.0f; bb[i] = ch; }
    v8f cz = {};
    v8f dd = __builtin_amdgcn_wmma_f32_16x16x32_f16(false, aa, false, bb,
                                                    (short)0, cz, false, false);
    {
        int lane = tid & 31, wv = tid >> 5;
        if (lane < 16) s_wred[wv * 16 + lane] = dd[0];
    }

    // ---- deterministic fixed-order tree reductions for the float sums ----
    s_red[tid] = sl1_part; s_red2[tid] = cep_part;
    __syncthreads();
    for (int s = TPB / 2; s > 0; s >>= 1) {
        if (tid < s) { s_red[tid] += s_red[tid + s]; s_red2[tid] += s_red2[tid + s]; }
        __syncthreads();
    }
    float blk_sl1 = 0.f, blk_cep = 0.f;
    if (tid == 0) {
        blk_sl1 = s_red[0]; blk_cep = s_red2[0];
        float t = 0.f;
        for (int i = 0; i < (TPB / 32) * 16; ++i) t += s_wred[i];
        s_count = (int)(t * (1.0f / 16.0f) + 0.5f);
    }
    __syncthreads();
    const int npos = s_count;
    int K = 3 * npos; if (K > P_) K = P_;

    // ---- hard-negative mining: exact K-th largest via bitwise binary search ----
    float neg_total = 0.f;
    if (K > 0) {                         // uniform branch (K same for all threads)
        unsigned cur = 0u;
        for (int bit = 30; bit >= 0; --bit) {
            if (tid == 0) s_cnt = 0;
            __syncthreads();
            unsigned cand = cur | (1u << bit);
            float t = __uint_as_float(cand);
            int lc = 0;
            for (int p = tid; p < P_; p += TPB) lc += (s_ce[p] >= t) ? 1 : 0;
            if (lc) atomicAdd(&s_cnt, lc);
            __syncthreads();
            if (s_cnt >= K) cur = cand;  // uniform update
            __syncthreads();
        }
        float vK = __uint_as_float(cur);
        if (tid == 0) s_cnt = 0;
        __syncthreads();
        float lsum = 0.f; int lc = 0;
        for (int p = tid; p < P_; p += TPB) {
            float v = s_ce[p];
            if (v > vK) { lsum += v; ++lc; }
        }
        if (lc) atomicAdd(&s_cnt, lc);
        s_red[tid] = lsum;
        __syncthreads();
        for (int s = TPB / 2; s > 0; s >>= 1) {
            if (tid < s) s_red[tid] += s_red[tid + s];
            __syncthreads();
        }
        if (tid == 0) neg_total = s_red[0] + (float)(K - s_cnt) * vK;
    }

    if (tid == 0) {
        ws_sl1[b] = blk_sl1; ws_cep[b] = blk_cep;
        ws_neg[b] = neg_total; ws_npos[b] = npos;
    }
}

__global__ __launch_bounds__(B_) void mbox_finalize(
    const float* __restrict__ ws_sl1, const float* __restrict__ ws_cep,
    const float* __restrict__ ws_neg, const int* __restrict__ ws_npos,
    float* __restrict__ out)
{
    __shared__ float r0[B_], r1[B_], r2[B_];
    __shared__ int   r3[B_];
    int t = threadIdx.x;
    r0[t] = ws_sl1[t]; r1[t] = ws_cep[t]; r2[t] = ws_neg[t]; r3[t] = ws_npos[t];
    __syncthreads();
    for (int s = B_ / 2; s > 0; s >>= 1) {
        if (t < s) { r0[t]+=r0[t+s]; r1[t]+=r1[t+s]; r2[t]+=r2[t+s]; r3[t]+=r3[t+s]; }
        __syncthreads();
    }
    if (t == 0) {
        float npt = (float)r3[0];
        float conf_loss = (r2[0] + r1[0]) / (npt + 1e-7f);
        float loc_loss  = r0[0] / (4.0f * npt);
        out[0] = conf_loss + loc_loss;
    }
}

extern "C" void kernel_launch(void* const* d_in, const int* in_sizes, int n_in,
                              void* d_out, int out_size, void* d_ws, size_t ws_size,
                              hipStream_t stream) {
    (void)in_sizes; (void)n_in; (void)out_size; (void)ws_size;
    const float* loc_preds  = (const float*)d_in[0];
    const float* conf_preds = (const float*)d_in[1];
    const float* boxes      = (const float*)d_in[2];
    const int*   labels     = (const int*)d_in[3];
    const float* priors     = (const float*)d_in[4];
    float* out = (float*)d_out;

    float* ws_sl1  = (float*)d_ws;            // [B]
    float* ws_cep  = ws_sl1 + B_;             // [B]
    float* ws_neg  = ws_cep + B_;             // [B]
    int*   ws_npos = (int*)(ws_neg + B_);     // [B]

    mbox_per_image<<<B_, TPB, 0, stream>>>(loc_preds, conf_preds, boxes, labels,
                                           priors, ws_sl1, ws_cep, ws_neg, ws_npos);
    mbox_finalize<<<1, B_, 0, stream>>>(ws_sl1, ws_cep, ws_neg, ws_npos, out);
}
